// RNNWithSoftmaxVectorSelection_54425825575655
// MI455X (gfx1250) — compile-verified, matrix operand order checked by reading, not run on hardware
//
#include <hip/hip_runtime.h>
#include <hip/hip_bf16.h>
#include <math.h>

// Problem constants (from reference): B=32, T=128, E=256, S=5, V=32000, A=2S=10
#define B_ 32
#define T_ 128
#define E_ 256
#define S_ 5
#define V_ 32000
#define A_ 10

#define VPARTS 8
#define VSPAN (V_ / VPARTS)      // 4000 columns per partition
#define TROWS 32                 // Wv rows (logit columns) per TDM tile
#define NTILES (VSPAN / TROWS)   // 125 tiles per partition
#define BROW 264                 // padded LDS row pitch (elements) = 256 + 8

typedef __attribute__((ext_vector_type(16))) __bf16 v16bf;
typedef __attribute__((ext_vector_type(8)))  float  v8f;
typedef unsigned int u32x4 __attribute__((ext_vector_type(4)));
typedef int          i32x4 __attribute__((ext_vector_type(4)));
typedef int          i32x8 __attribute__((ext_vector_type(8)));

union FragAB { v16bf v; uint4 q[2]; };
union FragC  { v8f v; float f[8]; };

__device__ __forceinline__ unsigned short f32_to_bf16(float f) {
  union { float f; unsigned int u; } c; c.f = f;
  unsigned int u = c.u;
  return (unsigned short)((u + 0x7FFFu + ((u >> 16) & 1u)) >> 16); // RNE
}
__device__ __forceinline__ float bf16_to_f32(unsigned short h) {
  union { unsigned int u; float f; } c; c.u = ((unsigned int)h) << 16;
  return c.f;
}

// ---------------------------------------------------------------------------
// TDM: load one 32x256 bf16 tile of Wv (rows v0..v0+31, all K) into LDS.
// 2-D D# per CDNA5 ISA:
//   data_size=1 (2B), tile_dim0=256, tile_dim1=32, tensor_dim0_stride=256,
//   pad_enable: +4 DWORDs every 128 DWORDs -> LDS row pitch 264 elements.
// clang-23 toolchain: 6-arg builtin (g0, g1, g2, g3, extra, cpol).
// ---------------------------------------------------------------------------
__device__ __forceinline__ void tdm_load_tile(const unsigned short* gsrc,
                                              unsigned int lds_off) {
  unsigned long long ga = (unsigned long long)(uintptr_t)gsrc;
  u32x4 g0;
  g0[0] = 1u;                                       // count=1 (valid user D#)
  g0[1] = lds_off;                                  // lds_addr (bytes)
  g0[2] = (unsigned int)ga;                         // global_addr[31:0]
  g0[3] = (unsigned int)((ga >> 32) & 0x1FFFFFFull) // global_addr[56:32]
        | (2u << 30);                               // type=2 ("image")
  i32x8 g1;
  g1[0] = (1 << 16)      // data_size = 2 bytes
        | (1 << 20)      // pad_enable
        | (6 << 22)      // pad_interval: every 128 DWORDs (one 512B row)
        | (3 << 25);     // pad_amount: 4 DWORDs (16B) -> pitch 264 elems
  g1[1] = (int)(256u << 16);               // tensor_dim0[15:0]
  g1[2] = (int)((32000u & 0xFFFFu) << 16); // tensor_dim1[15:0]
  g1[3] = (int)(256u << 16);               // tile_dim0 = 256
  g1[4] = TROWS;                           // tile_dim1 = 32, tile_dim2 = 0
  g1[5] = 256;                             // tensor_dim0_stride[31:0]
  g1[6] = 0;
  g1[7] = 0;
  i32x4 z4 = {0, 0, 0, 0};                 // groups 2/3 unused (2-D tensor)
  i32x8 z8 = {0, 0, 0, 0, 0, 0, 0, 0};
  __builtin_amdgcn_tensor_load_to_lds(g0, g1, z4, z4, z8, 0);
}

// ---------------------------------------------------------------------------
// Kernel 1: sequential RNN scan. One workgroup per batch element, 256 threads
// (thread j owns vector component j). Wt transposed+padded in LDS (263KB,
// dynamic) so the per-step GEMV is bank-conflict-free. Emits zs rows in bf16.
// ---------------------------------------------------------------------------
__global__ __launch_bounds__(256) void rnn_kernel(
    const float* __restrict__ latent, const float* __restrict__ shared_kv,
    const float* __restrict__ We, const float* __restrict__ be,
    const float* __restrict__ Wt, const float* __restrict__ bt,
    const float* __restrict__ noise, const int* __restrict__ zi,
    unsigned short* __restrict__ zs_bf16)
{
  extern __shared__ float smem[];
  float* wtT  = smem;                 // [256][257] Wt transposed, padded
  float* cand = wtT + 256 * 257;      // [10][256] candidate rows
  float* red  = cand + A_ * E_;       // [256] reduce scratch (sum)
  float* red2 = red + 256;            // [256] reduce scratch (sumsq)
  float* attL = red2 + 256;           // [16] attention logits/weights

  const int tid  = threadIdx.x;
  const int lane = tid & 31;
  const int wave = tid >> 5;
  const int b    = blockIdx.x;

  for (int idx = tid; idx < E_ * E_; idx += 256) {
    int j = idx >> 8, k = idx & 255;
    wtT[k * 257 + j] = Wt[idx];
  }
  for (int i = tid; i < S_ * E_; i += 256) cand[5 * E_ + i] = shared_kv[i];

  const float* zbase = latent + (size_t)zi[b] * (S_ * E_);
  float z = 0.0f;
  for (int s = 0; s < S_; ++s) {
    float x = zbase[s * E_ + tid];
    red[tid] = x; red2[tid] = x * x; __syncthreads();
    for (int off = 128; off > 0; off >>= 1) {
      if (tid < off) { red[tid] += red[tid + off]; red2[tid] += red2[tid + off]; }
      __syncthreads();
    }
    float mean = red[0] * (1.0f / E_);
    float var  = (red2[0] - (float)E_ * mean * mean) * (1.0f / (E_ - 1));
    float sd   = sqrtf(fmaxf(var, 0.0f));
    __syncthreads();
    float xn = x / (1e-5f + sd) * 0.113f
             + (noise[((size_t)b * S_ + s) * E_ + tid] - 0.5f) * 0.2f;
    if (s == 0) z = xn; else cand[s * E_ + tid] = xn;
  }
  __syncthreads();

  for (int t = 0; t < T_; ++t) {
    red[tid] = z; red2[tid] = z * z; __syncthreads();
    for (int off = 128; off > 0; off >>= 1) {
      if (tid < off) { red[tid] += red[tid + off]; red2[tid] += red2[tid + off]; }
      __syncthreads();
    }
    float mean = red[0] * (1.0f / E_);
    float var  = (red2[0] - (float)E_ * mean * mean) * (1.0f / (E_ - 1));
    float sd   = sqrtf(fmaxf(var, 0.0f));
    float zn   = z / (1e-5f + sd) * 0.113f;
    __syncthreads();
    cand[tid] = zn;
    __syncthreads();

    for (int a = wave; a < A_; a += 8) {
      float p = 0.0f;
      for (int i = lane; i < E_; i += 32) p += cand[i] * We[a * E_ + i];
      for (int off = 16; off > 0; off >>= 1) p += __shfl_down(p, off, 32);
      if (lane == 0) attL[a] = p + be[a];
    }
    __syncthreads();
    if (tid == 0) {
      float m = attL[0];
      for (int a = 1; a < A_; ++a) m = fmaxf(m, attL[a]);
      float ssum = 0.0f;
      for (int a = 0; a < A_; ++a) { float e = __expf(attL[a] - m); attL[a] = e; ssum += e; }
      float inv = 1.0f / ssum;
      for (int a = 0; a < A_; ++a) attL[a] *= inv;
    }
    __syncthreads();

    float emit = 0.0f;
    for (int a = 0; a < A_; ++a) emit += attL[a] * cand[a * E_ + tid];
    zs_bf16[((size_t)(b * T_ + t)) * E_ + tid] = f32_to_bf16(emit);

    float acc = 0.0f;
    for (int k = 0; k < E_; ++k) acc += wtT[k * 257 + tid] * cand[k];
    z = (zn + acc + bt[tid]) * 0.5f;
    __syncthreads();
  }
}

// ---------------------------------------------------------------------------
// Kernel 2: Wv f32 -> bf16
// ---------------------------------------------------------------------------
__global__ void cvt_bf16_kernel(const float* __restrict__ src,
                                unsigned short* __restrict__ dst, int n) {
  int i = blockIdx.x * blockDim.x + threadIdx.x;
  int stride = gridDim.x * blockDim.x;
  for (; i < n; i += stride) dst[i] = f32_to_bf16(src[i]);
}

// ---------------------------------------------------------------------------
// Kernel 3: partial logits GEMM + register-resident online logsumexp.
// Grid = 32 row-groups x 8 V-partitions. Each block: 128 rows (8 waves x 16,
// A fragments in 64 VGPRs/wave) x 4000 columns. 32x256 bf16 B tiles (16KB)
// are TDM-loaded into double-buffered LDS by wave 0 and shared by all waves:
// one tensor op + 2 barriers per 16 WMMAs/wave; every Wv byte feeds 128 rows.
// ---------------------------------------------------------------------------
__global__ __launch_bounds__(256) void logits_partial_kernel(
    const unsigned short* __restrict__ zs, const unsigned short* __restrict__ Wv,
    const float* __restrict__ bv, float* __restrict__ pM, float* __restrict__ pS)
{
  __shared__ __align__(16) unsigned short buf[2][TROWS * BROW];

  const int tid  = threadIdx.x;
  const int lane = tid & 31;
  const int wave = tid >> 5;
  const int rg   = blockIdx.x >> 3;     // row group 0..31
  const int vp   = blockIdx.x & 7;      // V partition 0..7
  const int R0   = rg * 128 + wave * 16;
  const int vstart = vp * VSPAN;

  const int m     = lane & 15;          // A row / B column within 16-col group
  const int khalf = (lane >> 4) * 8;    // ISA 16-bit operand lane split

  // A fragments: this wave's 16 rows x K=256 (two 16B chunks per K-block)
  FragAB afrag[8];
  {
    const unsigned short* arow = zs + (size_t)(R0 + m) * E_;
#pragma unroll
    for (int kc = 0; kc < 8; ++kc) {
      int k0 = kc * 32 + khalf;
      afrag[kc].q[0] = *(const uint4*)(arow + k0);
      afrag[kc].q[1] = *(const uint4*)(arow + k0 + 16);
    }
  }

  // per-lane-slot online logsumexp state (row = half*8 + i, columns 32it+g16+m)
  float M[8], S[8];
#pragma unroll
  for (int i = 0; i < 8; ++i) { M[i] = -INFINITY; S[i] = 0.0f; }

  if (wave == 0)
    tdm_load_tile(Wv + (size_t)vstart * E_, (unsigned int)(uintptr_t)&buf[0][0]);

  for (int it = 0; it < NTILES; ++it) {
    if (wave == 0) {
      if (it + 1 < NTILES) {
        tdm_load_tile(Wv + (size_t)(vstart + (it + 1) * TROWS) * E_,
                      (unsigned int)(uintptr_t)&buf[(it + 1) & 1][0]);
        __builtin_amdgcn_s_wait_tensorcnt((short)1);  // tile `it` complete
      } else {
        __builtin_amdgcn_s_wait_tensorcnt((short)0);
      }
    }
    __syncthreads();   // publish buf[it&1] to all waves

    const unsigned short* tilebase = &buf[it & 1][0];
#pragma unroll
    for (int g = 0; g < 2; ++g) {       // two 16-column groups per tile
      const unsigned short* brow = tilebase + (g * 16 + m) * BROW;
      // preload all 8 B fragments, then run the dense WMMA chain
      FragAB bfrag[8];
#pragma unroll
      for (int kc = 0; kc < 8; ++kc) {
        int k0 = kc * 32 + khalf;
        bfrag[kc].q[0] = *(const uint4*)(brow + k0);      // ds_load_b128
        bfrag[kc].q[1] = *(const uint4*)(brow + k0 + 16);
      }
      FragC acc;
#pragma unroll
      for (int i = 0; i < 8; ++i) acc.f[i] = 0.0f;
#pragma unroll
      for (int kc = 0; kc < 8; ++kc)
        acc.v = __builtin_amdgcn_wmma_f32_16x16x32_bf16(
            false, afrag[kc].v, false, bfrag[kc].v, (short)0, acc.v, false, false);

      float bvv = bv[vstart + it * TROWS + g * 16 + m];
#pragma unroll
      for (int i = 0; i < 8; ++i) {                       // online (M,S) update
        float x  = acc.f[i] + bvv;
        float Mn = fmaxf(M[i], x);
        S[i] = S[i] * __expf(M[i] - Mn) + __expf(x - Mn);
        M[i] = Mn;
      }
    }
    __syncthreads();   // all waves done with buf[it&1] before it is rewritten
  }

  // merge the 16 lanes of each half per row slot, write (M,S) partials
#pragma unroll
  for (int i = 0; i < 8; ++i) {
    float Mi = M[i], Si = S[i];
#pragma unroll
    for (int off = 8; off > 0; off >>= 1) {
      float Mo = __shfl_xor(Mi, off, 16);
      float So = __shfl_xor(Si, off, 16);
      float Mn = fmaxf(Mi, Mo);
      Si = Si * __expf(Mi - Mn) + So * __expf(Mo - Mn);
      Mi = Mn;
    }
    if ((lane & 15) == 0) {
      int row = R0 + (lane >> 4) * 8 + i;
      pM[row * VPARTS + vp] = Mi;
      pS[row * VPARTS + vp] = Si;
    }
  }
}

// ---------------------------------------------------------------------------
// Kernel 4: merge V-partition partials, add target logit, emit log-prob.
// 16 threads per row.
// ---------------------------------------------------------------------------
__global__ __launch_bounds__(256) void combine_kernel(
    const unsigned short* __restrict__ zs, const unsigned short* __restrict__ Wv,
    const float* __restrict__ bv, const int* __restrict__ y,
    const float* __restrict__ pM, const float* __restrict__ pS,
    float* __restrict__ out)
{
  const int tid = threadIdx.x;
  const int r = tid >> 4, ct = tid & 15;
  const int R = blockIdx.x * 16 + r;      // R = b*T + t; y is (B,T) row-major

  float M = -INFINITY, S = 0.0f;
  for (int p = 0; p < VPARTS; ++p) {
    float Mp = pM[R * VPARTS + p], Sp = pS[R * VPARTS + p];
    float Mn = fmaxf(M, Mp);
    S = S * __expf(M - Mn) + Sp * __expf(Mp - Mn);
    M = Mn;
  }

  int yv = y[R];
  const unsigned short* zr = zs + (size_t)R  * E_;
  const unsigned short* wr = Wv + (size_t)yv * E_;
  float d = 0.0f;
  for (int i = 0; i < 16; ++i) {
    int k = ct * 16 + i;
    d += bf16_to_f32(zr[k]) * bf16_to_f32(wr[k]);
  }
  for (int off = 8; off > 0; off >>= 1) d += __shfl_xor(d, off, 16);
  if (ct == 0) out[R] = (d + bv[yv]) - (M + __logf(S));
}

// ---------------------------------------------------------------------------
extern "C" void kernel_launch(void* const* d_in, const int* in_sizes, int n_in,
                              void* d_out, int out_size, void* d_ws, size_t ws_size,
                              hipStream_t stream) {
  const float* latent    = (const float*)d_in[0];
  const float* shared_kv = (const float*)d_in[1];
  const float* We        = (const float*)d_in[2];
  const float* be        = (const float*)d_in[3];
  const float* Wt        = (const float*)d_in[4];
  const float* bt        = (const float*)d_in[5];
  const float* Wv        = (const float*)d_in[6];
  const float* bv        = (const float*)d_in[7];
  const float* noise     = (const float*)d_in[8];
  const int*   zi        = (const int*)d_in[9];
  const int*   yy        = (const int*)d_in[10];
  float* out = (float*)d_out;

  // Workspace: [Wv bf16 16.384MB][zs bf16 2.097MB][pM 128KB][pS 128KB]
  unsigned short* Wv_bf16 = (unsigned short*)d_ws;
  unsigned short* zs_bf16 = Wv_bf16 + (size_t)V_ * E_;
  float* pM = (float*)(zs_bf16 + (size_t)B_ * T_ * E_);
  float* pS = pM + (size_t)B_ * T_ * VPARTS;

  cvt_bf16_kernel<<<2048, 256, 0, stream>>>(Wv, Wv_bf16, V_ * E_);

  size_t smem = (size_t)(256 * 257 + A_ * E_ + 256 + 256 + 16) * sizeof(float);
  rnn_kernel<<<B_, 256, smem, stream>>>(latent, shared_kv, We, be, Wt, bt,
                                        noise, zi, zs_bf16);

  logits_partial_kernel<<<256, 256, 0, stream>>>(zs_bf16, Wv_bf16, bv, pM, pS);
  combine_kernel<<<(B_ * T_) / 16, 256, 0, stream>>>(zs_bf16, Wv_bf16, bv, yy,
                                                     pM, pS, out);
}
